// GroupedQueryAttention_79474074845651
// MI455X (gfx1250) — matrix operand order repaired and checked by reading, not compiled, not run on hardware
//
#include <hip/hip_runtime.h>
#include <hip/hip_bf16.h>

// ---------------- problem constants (from reference) ----------------
#define BB   2
#define SS   2048
#define DD   1024
#define HH   16
#define HKV  4
#define DK   64
#define NREP (HH / HKV)

typedef __bf16 bf16_t;
typedef __attribute__((ext_vector_type(16))) __bf16 v16bf;
typedef __attribute__((ext_vector_type(8)))  float  v8f;
typedef __attribute__((ext_vector_type(4)))  int    v4i;

union FragBF { v16bf v; v4i q[2]; };

// ---- WMMA fragment loaders (16-bit layouts per CDNA5 ISA 7.12.2) ----
// A (16x32, row-major, K contiguous): lanes 0-15 row M=lane, K={0..7,16..23};
// lanes 16-31 same rows, K={8..15,24..31}. Two 16B loads per lane.
__device__ __forceinline__ v16bf load_a_frag(const bf16_t* base, int ld) {
  int lane = threadIdx.x & 31;
  int row  = lane & 15;
  int kb   = (lane < 16) ? 0 : 8;
  const bf16_t* p = base + (size_t)row * ld + kb;
  FragBF f;
  f.q[0] = *(const v4i*)(p);        // K = kb .. kb+7
  f.q[1] = *(const v4i*)(p + 16);   // K = kb+16 .. kb+23
  return f.v;
}

// B (32x16) supplied as B^T (N-major, K contiguous): lane holds column n=lane&15;
// lanes 0-15 K={0..7,16..23}, lanes 16-31 K={8..15,24..31} (two 16B chunks).
__device__ __forceinline__ v16bf load_b_frag(const bf16_t* baseT, int ld) {
  int lane = threadIdx.x & 31;
  int col  = lane & 15;
  int kb   = (lane < 16) ? 0 : 8;
  const bf16_t* p = baseT + (size_t)col * ld + kb;
  FragBF f;
  f.q[0] = *(const v4i*)(p);
  f.q[1] = *(const v4i*)(p + 16);
  return f.v;
}

__device__ __forceinline__ v8f wmma_bf16(v16bf a, v16bf b, v8f c) {
  return __builtin_amdgcn_wmma_f32_16x16x32_bf16(false, a, false, b, (short)0, c,
                                                 false, false);
}

// ---- DPP cross-lane reductions within each 16-lane DPP row (pure VALU) ----
template <int CTRL>
__device__ __forceinline__ float dpp_f32(float x) {
  int r = __builtin_amdgcn_update_dpp(0, __float_as_int(x), CTRL, 0xF, 0xF, true);
  return __int_as_float(r);
}
__device__ __forceinline__ float row_max16(float x) {
  x = fmaxf(x, dpp_f32<0xB1>(x));    // quad_perm(1,0,3,2)
  x = fmaxf(x, dpp_f32<0x4E>(x));    // quad_perm(2,3,0,1)
  x = fmaxf(x, dpp_f32<0x141>(x));   // row_half_mirror
  x = fmaxf(x, dpp_f32<0x140>(x));   // row_mirror
  return x;
}
__device__ __forceinline__ float row_sum16(float x) {
  x += dpp_f32<0xB1>(x);
  x += dpp_f32<0x4E>(x);
  x += dpp_f32<0x141>(x);
  x += dpp_f32<0x140>(x);
  return x;
}

// ---------------- generic bf16 WMMA GEMM ----------------
// C(MxN, f32) = A(MxK) * Bw(KxN, f32 weights). A element type templated
// (f32 activations or bf16 ctx); both converted to bf16 in LDS staging.
// Block: 256 threads = 8 waves; block tile 128(M) x 128(N); K step 32.
// Wave tile 32x64: 2 A-frags x 4 B-frags = 8 WMMA per K step.
#define ASTR 40
#define BSTR 40
template <typename AT>
__global__ __launch_bounds__(256) void gemm_bf16_wmma(const AT* __restrict__ A,
                                                      const float* __restrict__ Bw,
                                                      float* __restrict__ C,
                                                      int M, int N, int K) {
  __shared__ __align__(16) bf16_t As[128 * ASTR];
  __shared__ __align__(16) bf16_t Bs[128 * BSTR];
  const int tid  = threadIdx.x;
  const int wave = tid >> 5;
  const int lane = tid & 31;
  const int wm = (wave >> 1) * 32;   // wave row offset in block tile
  const int wn = (wave & 1) * 64;    // wave col offset in block tile
  const int m0 = blockIdx.y * 128;
  const int n0 = blockIdx.x * 128;

  v8f acc[2][4] = {};

  for (int k0 = 0; k0 < K; k0 += 32) {
    if (k0 + 32 < K) {  // hint next A tile into cache (global_prefetch_b8)
      __builtin_prefetch(A + (size_t)(m0 + (tid >> 1)) * K + k0 + 32, 0, 1);
    }
    // stage A: 128 rows x 32 k; 8 contiguous elements per thread x 2 passes
    #pragma unroll
    for (int i = 0; i < 2; ++i) {
      int c = tid + i * 256;          // 0..511
      int row = c >> 2;
      int kk  = (c & 3) * 8;
      const AT* src = A + (size_t)(m0 + row) * K + k0 + kk;
      bf16_t* dst = As + row * ASTR + kk;
      #pragma unroll
      for (int e = 0; e < 8; ++e) dst[e] = (bf16_t)(float)src[e];
    }
    // stage B transposed: Bs[n][k] <- Bw[k0+k][n0+n]; float4 loads along n
    #pragma unroll
    for (int i = 0; i < 4; ++i) {
      int c = tid + i * 256;          // 0..1023 float4 chunks
      int kk = c >> 5;                // 0..31
      int n4 = (c & 31) * 4;          // 0..124
      float4 v = *(const float4*)(Bw + (size_t)(k0 + kk) * N + n0 + n4);
      Bs[(n4 + 0) * BSTR + kk] = (bf16_t)v.x;
      Bs[(n4 + 1) * BSTR + kk] = (bf16_t)v.y;
      Bs[(n4 + 2) * BSTR + kk] = (bf16_t)v.z;
      Bs[(n4 + 3) * BSTR + kk] = (bf16_t)v.w;
    }
    __syncthreads();

    v16bf a0 = load_a_frag(As + wm * ASTR, ASTR);
    v16bf a1 = load_a_frag(As + (wm + 16) * ASTR, ASTR);
    #pragma unroll
    for (int j = 0; j < 4; ++j) {
      v16bf b = load_b_frag(Bs + (wn + j * 16) * BSTR, BSTR);
      acc[0][j] = wmma_bf16(a0, b, acc[0][j]);
      acc[1][j] = wmma_bf16(a1, b, acc[1][j]);
    }
    __syncthreads();
  }

  // C layout: lane holds column n=lane&15; VGPR r holds row r (+8 for lanes 16-31)
  const int col = lane & 15;
  const int rb  = (lane < 16) ? 0 : 8;
  #pragma unroll
  for (int i = 0; i < 2; ++i)
    #pragma unroll
    for (int j = 0; j < 4; ++j)
      #pragma unroll
      for (int r = 0; r < 8; ++r)
        C[(size_t)(m0 + wm + i * 16 + rb + r) * N + n0 + wn + j * 16 + col] =
            acc[i][j][r];
}

// ---------------- RoPE (note: reference rotates by HEAD index, not seq pos) ----------------
__global__ __launch_bounds__(256) void rope_q_kernel(const float* __restrict__ Qf,
                                                     bf16_t* __restrict__ Qb) {
  size_t idx = (size_t)blockIdx.x * blockDim.x + threadIdx.x;  // pair index
  const size_t total = (size_t)BB * SS * HH * (DK / 2);
  if (idx >= total) return;
  int j = (int)(idx % (DK / 2));
  size_t t = idx / (DK / 2);
  int h = (int)(t % HH);
  size_t tok = t / HH;                 // b*S + s
  const float* p = Qf + tok * (HH * DK) + h * DK + 2 * j;
  float x1 = p[0], x2 = p[1];
  float ang = (float)h * __powf(10000.f, -(2.f * j) / (float)DK);
  float c = __cosf(ang), s = __sinf(ang);
  float o1 = (x1 * c - x2 * s) * 0.125f;   // fold 1/sqrt(DK)
  float o2 = (x1 * s + x2 * c) * 0.125f;
  int b = (int)(tok / SS), sq = (int)(tok % SS);
  size_t o = (((size_t)b * HH + h) * SS + sq) * DK + 2 * j;
  Qb[o] = (bf16_t)o1;
  Qb[o + 1] = (bf16_t)o2;
}

__global__ __launch_bounds__(256) void rope_k_kernel(const float* __restrict__ Kf,
                                                     bf16_t* __restrict__ Kb) {
  size_t idx = (size_t)blockIdx.x * blockDim.x + threadIdx.x;
  const size_t total = (size_t)BB * SS * HKV * (DK / 2);
  if (idx >= total) return;
  int j = (int)(idx % (DK / 2));
  size_t t = idx / (DK / 2);
  int g = (int)(t % HKV);
  size_t tok = t / HKV;
  const float* p = Kf + tok * (HKV * DK) + g * DK + 2 * j;
  float x1 = p[0], x2 = p[1];
  float ang = (float)g * __powf(10000.f, -(2.f * j) / (float)DK);
  float c = __cosf(ang), s = __sinf(ang);
  int b = (int)(tok / SS), sq = (int)(tok % SS);
  size_t o = (((size_t)b * HKV + g) * SS + sq) * DK + 2 * j;
  Kb[o] = (bf16_t)(x1 * c - x2 * s);
  Kb[o + 1] = (bf16_t)(x1 * s + x2 * c);
}

__global__ __launch_bounds__(256) void v_transpose_kernel(const float* __restrict__ Vf,
                                                          bf16_t* __restrict__ VT) {
  size_t idx = (size_t)blockIdx.x * blockDim.x + threadIdx.x;
  const size_t total = (size_t)BB * SS * HKV * DK;
  if (idx >= total) return;
  int col = (int)(idx % (HKV * DK));
  size_t tok = idx / (HKV * DK);
  int g = col / DK, d = col % DK;
  int b = (int)(tok / SS), sq = (int)(tok % SS);
  VT[(((size_t)b * HKV + g) * DK + d) * SS + sq] = (bf16_t)Vf[idx];
}

// ---------------- flash attention (causal, GQA) ----------------
// 4 waves / block, each wave owns one 16-row q tile; online softmax with
// DPP row-max; per-lane partial row sums reduced once in the epilogue.
#define PSTR 40
__global__ __launch_bounds__(128) void attn_kernel(const bf16_t* __restrict__ Qb,
                                                   const bf16_t* __restrict__ Kb,
                                                   const bf16_t* __restrict__ VT,
                                                   bf16_t* __restrict__ ctx) {
  __shared__ __align__(16) bf16_t Ps[4][16 * PSTR];
  const int wave = threadIdx.x >> 5;
  const int lane = threadIdx.x & 31;
  const int qt = blockIdx.x * 4 + wave;
  const int bh = blockIdx.y;
  const int b = bh / HH, h = bh % HH, g = h / NREP;
  const int q0 = qt * 16;

  const bf16_t* Qbase = Qb + (((size_t)b * HH + h) * SS + q0) * DK;
  const bf16_t* Kbase = Kb + (((size_t)b * HKV + g) * SS) * DK;
  const bf16_t* Vbase = VT + (((size_t)b * HKV + g) * DK) * SS;

  // Q fragments for dk 0..31 and 32..63 (held across whole kv loop)
  const v16bf qf0 = load_a_frag(Qbase, DK);
  const v16bf qf1 = load_a_frag(Qbase + 32, DK);

  const float NEG = -1.0e30f;
  float mrow[8], lsum[8];
  #pragma unroll
  for (int r = 0; r < 8; ++r) { mrow[r] = NEG; lsum[r] = 0.f; }
  v8f acc[4] = {v8f{}, v8f{}, v8f{}, v8f{}};

  const int col_l = lane & 15;
  const int rb = (lane < 16) ? 0 : 8;
  const int kv_end = q0 + 16;  // causal

  for (int kv = 0; kv < kv_end; kv += 32) {
    // ---- scores: 16x32 = two C fragments (Q already scaled by 1/sqrt(DK)) ----
    v8f sc[2] = {v8f{}, v8f{}};
    #pragma unroll
    for (int j = 0; j < 2; ++j) {
      v16bf kb0 = load_b_frag(Kbase + (size_t)(kv + j * 16) * DK, DK);       // dk 0..31
      v16bf kb1 = load_b_frag(Kbase + (size_t)(kv + j * 16) * DK + 32, DK);  // dk 32..63
      sc[j] = wmma_bf16(qf0, kb0, sc[j]);
      sc[j] = wmma_bf16(qf1, kb1, sc[j]);
    }
    // ---- mask + online softmax (per-lane stats cover rows rb..rb+7) ----
    const int c0 = kv + col_l;
    #pragma unroll
    for (int r = 0; r < 8; ++r) {
      const int row = q0 + rb + r;
      float v0 = (c0 <= row) ? sc[0][r] : NEG;
      float v1 = (c0 + 16 <= row) ? sc[1][r] : NEG;
      const float mx = row_max16(fmaxf(v0, v1));       // DPP, no LDS
      const float mnew = fmaxf(mrow[r], mx);
      const float ef = __expf(mrow[r] - mnew);
      const float p0 = __expf(v0 - mnew);
      const float p1 = __expf(v1 - mnew);
      lsum[r] = lsum[r] * ef + p0 + p1;                // per-lane partial sum
      mrow[r] = mnew;
      Ps[wave][(rb + r) * PSTR + col_l] = (bf16_t)p0;
      Ps[wave][(rb + r) * PSTR + col_l + 16] = (bf16_t)p1;
      #pragma unroll
      for (int j = 0; j < 4; ++j) acc[j][r] *= ef;
    }
    // ---- PV: A = P (16x32 from LDS), B = V^T columns (contiguous over seq) ----
    v16bf pa = load_a_frag(Ps[wave], PSTR);
    #pragma unroll
    for (int j = 0; j < 4; ++j) {
      v16bf vb = load_b_frag(Vbase + (size_t)(j * 16) * SS + kv, SS);
      acc[j] = wmma_bf16(pa, vb, acc[j]);
    }
  }

  // ---- epilogue: reduce row sums once, normalize, write ctx[B][S][H*DK] ----
  float linv[8];
  #pragma unroll
  for (int r = 0; r < 8; ++r) linv[r] = 1.0f / row_sum16(lsum[r]);
  #pragma unroll
  for (int j = 0; j < 4; ++j) {
    #pragma unroll
    for (int r = 0; r < 8; ++r) {
      float o = acc[j][r] * linv[r];
      ctx[((size_t)b * SS + q0 + rb + r) * (HH * DK) + h * DK + j * 16 + col_l] =
          (bf16_t)o;
    }
  }
}

// ---------------- host launcher ----------------
extern "C" void kernel_launch(void* const* d_in, const int* in_sizes, int n_in,
                              void* d_out, int out_size, void* d_ws, size_t ws_size,
                              hipStream_t stream) {
  const float* x   = (const float*)d_in[0];
  const float* w_q = (const float*)d_in[1];
  const float* w_k = (const float*)d_in[2];
  const float* w_v = (const float*)d_in[3];
  const float* w_o = (const float*)d_in[4];
  // d_in[5] = mask (causal, known analytically; ignored)
  float* out = (float*)d_out;

  const size_t M = (size_t)BB * SS;          // 4096 tokens
  // workspace carve-up
  float*  Qf  = (float*)d_ws;                          // M x 1024 f32
  float*  Kf  = Qf + M * (HH * DK);                    // M x 256  f32
  float*  Vf  = Kf + M * (HKV * DK);                   // M x 256  f32
  bf16_t* Qb  = (bf16_t*)(Vf + M * (HKV * DK));        // [B][H][S][DK]
  bf16_t* Kb  = Qb + (size_t)BB * HH * SS * DK;        // [B][HKV][S][DK]
  bf16_t* VT  = Kb + (size_t)BB * HKV * SS * DK;       // [B][HKV][DK][S]
  bf16_t* ctx = VT + (size_t)BB * HKV * DK * SS;       // [B][S][H*DK]

  // 1) projections (bf16 WMMA, f32 accumulate)
  gemm_bf16_wmma<float><<<dim3((HH * DK) / 128, M / 128), 256, 0, stream>>>(
      x, w_q, Qf, (int)M, HH * DK, DD);
  gemm_bf16_wmma<float><<<dim3((HKV * DK) / 128, M / 128), 256, 0, stream>>>(
      x, w_k, Kf, (int)M, HKV * DK, DD);
  gemm_bf16_wmma<float><<<dim3((HKV * DK) / 128, M / 128), 256, 0, stream>>>(
      x, w_v, Vf, (int)M, HKV * DK, DD);

  // 2) rope + layout transforms (+ 1/sqrt(DK) folded into Q)
  {
    size_t n = M * HH * (DK / 2);
    rope_q_kernel<<<(unsigned)((n + 255) / 256), 256, 0, stream>>>(Qf, Qb);
  }
  {
    size_t n = M * HKV * (DK / 2);
    rope_k_kernel<<<(unsigned)((n + 255) / 256), 256, 0, stream>>>(Kf, Kb);
  }
  {
    size_t n = M * HKV * DK;
    v_transpose_kernel<<<(unsigned)((n + 255) / 256), 256, 0, stream>>>(Vf, VT);
  }

  // 3) causal flash attention, GQA (4 q-tiles per block, one per wave)
  attn_kernel<<<dim3(SS / (16 * 4), BB * HH), 128, 0, stream>>>(Qb, Kb, VT, ctx);

  // 4) output projection -> f32 d_out
  gemm_bf16_wmma<bf16_t><<<dim3(DD / 128, M / 128), 256, 0, stream>>>(
      ctx, w_o, out, (int)M, DD, DD);
}
